// ClassOverlapMaskFilter_89026082111536
// MI455X (gfx1250) — compile-verified, matrix operand order checked by reading, not run on hardware
//
#include <hip/hip_runtime.h>

// Problem geometry (fixed by the reference).
#define NMASK 16
#define HWDIM 768
#define PIX   (HWDIM * HWDIM)   // 589824 pixels per mask
#define NWRD  (PIX / 32)        // 18432 packed u32 words per mask row
#define NWRD4 (NWRD / 4)        // 4608 uint4 (128-bit) quads per mask row

static_assert(PIX % 32 == 0, "pixels must pack into u32 words");
static_assert(NWRD % 4 == 0, "words must pack into 128-bit quads");

// ---------------------------------------------------------------------------
// 0) zero the popcount accumulators (ws is poisoned by the harness)
// ---------------------------------------------------------------------------
__global__ void init_pc_kernel(float* __restrict__ pc) {
  if (threadIdx.x < 32) pc[threadIdx.x] = 0.0f;
}

// ---------------------------------------------------------------------------
// 1) bitpack: 32 lanes test (v > 0) -> one u32 word via wave32 ballot
// ---------------------------------------------------------------------------
__global__ void pack_kernel(const float* __restrict__ src,
                            unsigned* __restrict__ bits, int nwords) {
  const int lane  = threadIdx.x & 31;
  const int wave  = blockIdx.x * (blockDim.x >> 5) + (threadIdx.x >> 5);
  const int nwave = gridDim.x * (blockDim.x >> 5);
  for (int w = wave; w < nwords; w += nwave) {
    float v = src[(size_t)w * 32 + lane];
    unsigned m = __builtin_amdgcn_ballot_w32(v > 0.0f);
    if (lane == 0) bits[w] = m;
  }
}

// ---------------------------------------------------------------------------
// 2) initial per-row pixel counts as a 16xP * ones(P) GEMM via WMMA.
//    B == all-ones, so the K-permutation within a row is sum-invariant:
//    lane L (resp. L+16) supplies 16 contiguous f16 values of row L&15,
//    covering chunk K 0..15 (resp. 16..31). Every D column is identical,
//    so lanes 0 / 16 hold rowsums for M 0..7 / 8..15 in their 8 acc VGPRs.
// ---------------------------------------------------------------------------
typedef __attribute__((ext_vector_type(16))) _Float16 v16h;
typedef __attribute__((ext_vector_type(8)))  float    v8f;

__global__ void rowsum_wmma_kernel(const float* __restrict__ masks,
                                   float* __restrict__ pc) {
  const int lane  = threadIdx.x & 31;
  const int wave  = blockIdx.x * (blockDim.x >> 5) + (threadIdx.x >> 5);
  const int nwave = gridDim.x * (blockDim.x >> 5);
  const int row   = lane & 15;
  const int hi    = lane >> 4;   // 0: K 0..15 of chunk, 1: K 16..31

  v16h ones;
#pragma unroll
  for (int t = 0; t < 16; ++t) ones[t] = (_Float16)1.0f;

  v8f c = {};
  const int nchunk = PIX / 32;   // K-chunks of 32 per row
  for (int ch = wave; ch < nchunk; ch += nwave) {
    const float* s = masks + (size_t)row * PIX + ch * 32 + hi * 16;
    v16h a;
#pragma unroll
    for (int t = 0; t < 16; ++t) a[t] = (_Float16)s[t];  // exact: values are 0/1
    c = __builtin_amdgcn_wmma_f32_16x16x32_f16(false, a, false, ones,
                                               (short)0, c, false, false);
  }
  if ((lane & 15) == 0) {        // lanes 0 and 16: one column of D
#pragma unroll
    for (int v = 0; v < 8; ++v) atomicAdd(&pc[hi * 8 + v], c[v]);
  }
}

// ---------------------------------------------------------------------------
// 3) the sequential 256-pair filter. One 1024-thread workgroup; A-row cached
//    in LDS (73.7 KB of the 320 KB WGP LDS) for the whole j-sweep. Popcounts
//    maintained incrementally; only popc(a & b) is reduced per pair.
//    All row traffic is 128-bit (b128 global / ds) to minimize per-pair
//    latency on the 256-step serial chain.
// ---------------------------------------------------------------------------
__device__ __forceinline__ int popc4(uint4 v) {
  return __popc(v.x) + __popc(v.y) + __popc(v.z) + __popc(v.w);
}

__global__ __launch_bounds__(1024)
void filter_kernel(unsigned* __restrict__ Abits, unsigned* __restrict__ Bbits,
                   const float* __restrict__ scores_a,
                   const float* __restrict__ scores_b,
                   const float* __restrict__ pcAf, const float* __restrict__ pcBf,
                   float* __restrict__ keep) {
  extern __shared__ uint4 sA[];                 // NWRD4 quads (73728 B, dynamic)
  __shared__ int   s_red[32];
  __shared__ int   s_pcA[16], s_pcB[16];
  __shared__ unsigned char s_supA[16], s_supB[16];
  __shared__ float s_sa[16], s_sb[16];
  __shared__ int   s_dec;

  const int tid = threadIdx.x;
  if (tid < 16) {
    s_pcA[tid] = (int)(pcAf[tid] + 0.5f);       // exact integer float sums
    s_pcB[tid] = (int)(pcBf[tid] + 0.5f);
    s_supA[tid] = 0; s_supB[tid] = 0;
    s_sa[tid] = scores_a[tid]; s_sb[tid] = scores_b[tid];
  }
  __syncthreads();

  for (int i = 0; i < NMASK; ++i) {
    uint4* aRow = (uint4*)(Abits + (size_t)i * NWRD);
    for (int t = tid; t < NWRD4; t += 1024) sA[t] = aRow[t];
    __syncthreads();

    for (int j = 0; j < NMASK; ++j) {
      uint4* bRow = (uint4*)(Bbits + (size_t)j * NWRD);

      // prefetch next B row into cache while we reduce this one
      {
        const uint4* bn = (const uint4*)(Bbits + (size_t)((j + 1) & 15) * NWRD);
        int t = tid * 8;                         // one 128B line per thread
        if (t < NWRD4) __builtin_prefetch((const void*)(bn + t), 0, 1);
      }

      // intersection popcount reduction (128-bit granularity)
      int acc = 0;
      for (int t = tid; t < NWRD4; t += 1024) {
        uint4 a4 = sA[t];
        uint4 b4 = bRow[t];
        acc += __popc(a4.x & b4.x) + __popc(a4.y & b4.y) +
               __popc(a4.z & b4.z) + __popc(a4.w & b4.w);
      }
#pragma unroll
      for (int off = 16; off > 0; off >>= 1) acc += __shfl_down(acc, off);
      if ((tid & 31) == 0) s_red[tid >> 5] = acc;
      __syncthreads();

      if (tid == 0) {
        int inter = 0;
#pragma unroll
        for (int w = 0; w < 32; ++w) inter += s_red[w];
        int uni = s_pcA[i] + s_pcB[j] - inter;
        float iou = (float)inter / fmaxf((float)uni, 1.0f);
        bool a_wins = s_sa[i] > s_sb[j];
        int dec = 0;
        if (iou > 0.8f) {                        // suppress loser (flag only)
          if (a_wins) s_supB[j] = 1; else s_supA[i] = 1;
        } else if (iou > 0.0f) {                 // zero loser's intersection
          if (a_wins) { dec = 2; s_pcB[j] -= inter; }
          else        { dec = 1; s_pcA[i] -= inter; }
        }
        s_dec = dec;
      }
      __syncthreads();

      const int dec = s_dec;
      if (dec == 2) {                            // b loses: b &= ~a
        for (int t = tid; t < NWRD4; t += 1024) {
          uint4 a4 = sA[t];
          uint4 b4 = bRow[t];
          b4.x &= ~a4.x; b4.y &= ~a4.y; b4.z &= ~a4.z; b4.w &= ~a4.w;
          bRow[t] = b4;
        }
        __threadfence_block();
      } else if (dec == 1) {                     // a loses: a &= ~b (in LDS)
        for (int t = tid; t < NWRD4; t += 1024) {
          uint4 a4 = sA[t];
          uint4 b4 = bRow[t];
          a4.x &= ~b4.x; a4.y &= ~b4.y; a4.z &= ~b4.z; a4.w &= ~b4.w;
          sA[t] = a4;
        }
      }
      __syncthreads();
    }

    // write the (possibly mutated) A row back for the expand kernel
    for (int t = tid; t < NWRD4; t += 1024) aRow[t] = sA[t];
    __syncthreads();
  }

  if (tid < 16)      keep[tid] = s_supA[tid] ? 0.0f : 1.0f;
  else if (tid < 32) keep[tid] = s_supB[tid - 16] ? 0.0f : 1.0f;
}

// ---------------------------------------------------------------------------
// 4) expand bits -> float output [A | B | keep_a | keep_b], float4 stores
// ---------------------------------------------------------------------------
__global__ void expand_kernel(const unsigned* __restrict__ Abits,
                              const unsigned* __restrict__ Bbits,
                              const float* __restrict__ keep,
                              float* __restrict__ out) {
  const long long NP  = (long long)NMASK * PIX;
  const long long q   = (long long)blockIdx.x * blockDim.x + threadIdx.x;
  const long long g   = q * 4;                  // first of 4 consecutive pixels
  if (g < 2 * NP) {
    const unsigned* bits = Abits;
    long long h = g;
    int koff = 0;
    if (g >= NP) { bits = Bbits; h = g - NP; koff = 16; }
    int row = (int)(h / PIX);
    unsigned w = bits[(size_t)row * NWRD + (size_t)((h % PIX) >> 5)];
    float k = keep[koff + row];
    int sh = (int)h & 31;                       // h % 32, multiple of 4
    float4 o;
    o.x = (k != 0.0f && ((w >> (sh + 0)) & 1u)) ? 1.0f : 0.0f;
    o.y = (k != 0.0f && ((w >> (sh + 1)) & 1u)) ? 1.0f : 0.0f;
    o.z = (k != 0.0f && ((w >> (sh + 2)) & 1u)) ? 1.0f : 0.0f;
    o.w = (k != 0.0f && ((w >> (sh + 3)) & 1u)) ? 1.0f : 0.0f;
    *(float4*)(out + g) = o;
  }
  if (blockIdx.x == 0 && threadIdx.x < 32) out[2 * NP + threadIdx.x] = keep[threadIdx.x];
}

// ---------------------------------------------------------------------------
extern "C" void kernel_launch(void* const* d_in, const int* in_sizes, int n_in,
                              void* d_out, int out_size, void* d_ws, size_t ws_size,
                              hipStream_t stream) {
  (void)in_sizes; (void)n_in; (void)out_size; (void)ws_size;
  const float* masks_a  = (const float*)d_in[0];
  const float* masks_b  = (const float*)d_in[1];
  const float* scores_a = (const float*)d_in[2];
  const float* scores_b = (const float*)d_in[3];

  // workspace layout: Abits | Bbits | pcA(16f) | pcB(16f) | keep(32f)
  unsigned* Abits = (unsigned*)d_ws;
  unsigned* Bbits = Abits + (size_t)NMASK * NWRD;
  float*    pcA   = (float*)(Bbits + (size_t)NMASK * NWRD);
  float*    pcB   = pcA + 16;
  float*    keep  = pcB + 16;
  float*    out   = (float*)d_out;

  hipLaunchKernelGGL(init_pc_kernel, dim3(1), dim3(32), 0, stream, pcA);

  const int packWords = NMASK * NWRD;   // 294912 words per mask set
  hipLaunchKernelGGL(pack_kernel, dim3(576), dim3(256), 0, stream,
                     masks_a, Abits, packWords);
  hipLaunchKernelGGL(pack_kernel, dim3(576), dim3(256), 0, stream,
                     masks_b, Bbits, packWords);

  hipLaunchKernelGGL(rowsum_wmma_kernel, dim3(144), dim3(256), 0, stream,
                     masks_a, pcA);
  hipLaunchKernelGGL(rowsum_wmma_kernel, dim3(144), dim3(256), 0, stream,
                     masks_b, pcB);

  hipLaunchKernelGGL(filter_kernel, dim3(1), dim3(1024),
                     NWRD * sizeof(unsigned), stream,
                     Abits, Bbits, scores_a, scores_b, pcA, pcB, keep);

  const long long quads = (2LL * NMASK * PIX) / 4;   // 4,718,592 float4 stores
  hipLaunchKernelGGL(expand_kernel, dim3((unsigned)((quads + 255) / 256)),
                     dim3(256), 0, stream, Abits, Bbits, keep, out);
}
